// RGAT_9483287790176
// MI455X (gfx1250) — matrix operand order repaired and checked by reading, not compiled, not run on hardware
//
#include <hip/hip_runtime.h>
#include <hip/hip_bf16.h>
#include <stdint.h>

typedef __attribute__((ext_vector_type(16))) _Float16 v16h;
typedef __attribute__((ext_vector_type(8)))  _Float16 v8h;
typedef __attribute__((ext_vector_type(8)))  float    v8f;

#define NN   50000
#define FFE  128
#define HH   4
#define CCH  32
#define HC   128
#define EEDG 800000
#define RRE  8
#define DEE  64

#if __has_builtin(__builtin_amdgcn_global_load_async_to_lds_b128)
#define HAVE_ASYNC_LDS 1
typedef __attribute__((ext_vector_type(4))) int v4i;
typedef __attribute__((address_space(1))) v4i gas_v4i;   // global int4
typedef __attribute__((address_space(3))) v4i las_v4i;   // LDS int4
#endif

// ---- order-preserving float<->uint key for atomicMax on signed floats ----
__device__ __forceinline__ unsigned fkey(float f) {
  unsigned u = __float_as_uint(f);
  return (u & 0x80000000u) ? ~u : (u | 0x80000000u);
}
__device__ __forceinline__ float fkey_dec(unsigned k) {
  return (k & 0x80000000u) ? __uint_as_float(k ^ 0x80000000u)
                           : __uint_as_float(~k);
}

// ---- zero per-call accumulators (maxkey, denom) ----
__global__ void rgat_init_kernel(unsigned* __restrict__ maxkey,
                                 float* __restrict__ denom) {
  int i = blockIdx.x * blockDim.x + threadIdx.x;
  if (i < NN * HH) { maxkey[i] = 0u; denom[i] = 0.0f; }
}

// ---- e_rel[r, :] = edge_emb[r, :] @ We  (8 x 128, tiny) ----
__global__ void rgat_erel_kernel(const float* __restrict__ emb,
                                 const float* __restrict__ We,
                                 float* __restrict__ erel) {
  int i = blockIdx.x * blockDim.x + threadIdx.x;
  if (i >= RRE * HC) return;
  int r = i / HC, j = i % HC;
  float acc = 0.0f;
  for (int d = 0; d < DEE; ++d) acc += emb[r * DEE + d] * We[d * HC + j];
  erel[i] = acc;
}

// ---- transpose + convert the four weight matrices to f16: Wt[mi][n][k] ----
__global__ void rgat_wtconv_kernel(const float* __restrict__ Wq,
                                   const float* __restrict__ Wk,
                                   const float* __restrict__ Wv,
                                   const float* __restrict__ Wsk,
                                   _Float16* __restrict__ Wt) {
  int i = blockIdx.x * blockDim.x + threadIdx.x;   // 4*128*128 = 65536
  if (i >= 4 * HC * FFE) return;
  int mi = i >> 14;
  int n  = (i >> 7) & 127;
  int k  = i & 127;
  const float* Wm[4] = {Wq, Wk, Wv, Wsk};
  Wt[i] = (_Float16)Wm[mi][(size_t)k * HC + n];    // i == (mi*128+n)*128+k
}

// ---- fused projections: Q,K,V -> workspace; x@Wskip -> out (overwrite) ----
__global__ __launch_bounds__(256)
void rgat_proj_kernel(const float* __restrict__ x,
                      const _Float16* __restrict__ Wt,
                      float* __restrict__ Q, float* __restrict__ K,
                      float* __restrict__ V, float* __restrict__ Out) {
  __shared__ float xsf[16][FFE];                   // 8 KB fp32 x-tile
  const int rowbase = blockIdx.x * 16;

  { // stage 16x128 fp32 x-tile into LDS (async DMA if available)
    int t  = threadIdx.x;
    int r  = t >> 4;
    int c0 = (t & 15) * 8;
    const float* gp = x + (size_t)(rowbase + r) * FFE + c0;
    float*       lp = &xsf[r][c0];
#if defined(HAVE_ASYNC_LDS)
    __builtin_amdgcn_global_load_async_to_lds_b128((gas_v4i*)gp,
                                                   (las_v4i*)lp, 0, 0);
    __builtin_amdgcn_global_load_async_to_lds_b128((gas_v4i*)(gp + 4),
                                                   (las_v4i*)(lp + 4), 0, 0);
#if __has_builtin(__builtin_amdgcn_s_wait_asynccnt)
    __builtin_amdgcn_s_wait_asynccnt(0);
#else
    asm volatile("s_wait_asynccnt 0x0" ::: "memory");
#endif
#else
    *(float4*)lp       = *(const float4*)gp;
    *(float4*)(lp + 4) = *(const float4*)(gp + 4);
#endif
  }
  __syncthreads();

  const int lane = threadIdx.x & 31;
  const int wave = threadIdx.x >> 5;
  const int hf   = lane >> 4;     // 0 or 1 (lane half)
  const int l15  = lane & 15;

  // A fragments (ISA 16-bit A 16x32 layout: lanes 0-15 K={0..7,16..23}),
  // built from two contiguous float4 pairs per K-step -> ds_load_b128.
  v16h afrag[4];
  #pragma unroll
  for (int ki = 0; ki < 4; ++ki) {
    const float4* p0 = (const float4*)&xsf[l15][ki * 32 + 8 * hf];
    const float4* p1 = (const float4*)&xsf[l15][ki * 32 + 16 + 8 * hf];
    float4 a0 = p0[0], a1 = p0[1], a2 = p1[0], a3 = p1[1];
    afrag[ki][0]  = (_Float16)a0.x; afrag[ki][1]  = (_Float16)a0.y;
    afrag[ki][2]  = (_Float16)a0.z; afrag[ki][3]  = (_Float16)a0.w;
    afrag[ki][4]  = (_Float16)a1.x; afrag[ki][5]  = (_Float16)a1.y;
    afrag[ki][6]  = (_Float16)a1.z; afrag[ki][7]  = (_Float16)a1.w;
    afrag[ki][8]  = (_Float16)a2.x; afrag[ki][9]  = (_Float16)a2.y;
    afrag[ki][10] = (_Float16)a2.z; afrag[ki][11] = (_Float16)a2.w;
    afrag[ki][12] = (_Float16)a3.x; afrag[ki][13] = (_Float16)a3.y;
    afrag[ki][14] = (_Float16)a3.z; afrag[ki][15] = (_Float16)a3.w;
  }

  float* Om[4] = {Q, K, V, Out};

  #pragma unroll
  for (int t = 0; t < 4; ++t) {
    const int coltile = wave * 4 + t;             // 0..31 over 4 matrices
    const int mi      = coltile >> 3;             // which weight matrix
    const int ncol    = (coltile & 7) * 16 + l15; // column within matrix
    // B fragment: per lane K={8h..8h+7,16+8h..16+8h+7} of Wt[mi][ncol][.] (column of W)
    const _Float16* wb = Wt + ((size_t)(mi * HC + ncol)) * FFE;
    v8f c = {};
    #pragma unroll
    for (int ki = 0; ki < 4; ++ki) {
      const v8h* bp = (const v8h*)(wb + ki * 32 + 8 * hf);
      v8h b0 = bp[0], b1 = bp[2];                 // two global_load_b128
      v16h b;
      #pragma unroll
      for (int i = 0; i < 8; ++i) { b[i] = b0[i]; b[8 + i] = b1[i]; }
      c = __builtin_amdgcn_wmma_f32_16x16x32_f16(false, afrag[ki], false, b,
                                                 (short)0, c, false, false);
    }
    float* O = Om[mi];
    #pragma unroll
    for (int i = 0; i < 8; ++i) {
      int m = i + 8 * hf;  // C/D layout: VGPR i -> row i (+8 for upper lanes)
      O[(size_t)(rowbase + m) * HC + ncol] = c[i];
    }
  }
}

// ---- pass 1: logits per (edge, head); atomic segment-max (wave per edge) ----
__global__ __launch_bounds__(256)
void rgat_logits_kernel(const float* __restrict__ Q, const float* __restrict__ K,
                        const float* __restrict__ erel,
                        const long long* __restrict__ eidx,
                        const long long* __restrict__ etype,
                        float* __restrict__ logits,
                        unsigned* __restrict__ maxkey) {
  const int e    = blockIdx.x * 8 + (threadIdx.x >> 5);
  const int lane = threadIdx.x & 31;
  const long long src = eidx[e];
  const long long dst = eidx[EEDG + e];
  const long long rel = etype[e];
  const float4 qv = *(const float4*)(Q    + (size_t)dst * HC + lane * 4);
  const float4 kv = *(const float4*)(K    + (size_t)src * HC + lane * 4);
  const float4 ev = *(const float4*)(erel + (size_t)rel * HC + lane * 4);
  float p = qv.x * (kv.x + ev.x) + qv.y * (kv.y + ev.y) +
            qv.z * (kv.z + ev.z) + qv.w * (kv.w + ev.w);
  p += __shfl_xor(p, 1, 32);   // reduce within 8-lane head groups
  p += __shfl_xor(p, 2, 32);
  p += __shfl_xor(p, 4, 32);
  if ((lane & 7) == 0) {
    int h = lane >> 3;
    float lg = p * 0.17677669529663688f;  // 1/sqrt(C)
    logits[(size_t)e * HH + h] = lg;
    atomicMax(&maxkey[(size_t)dst * HH + h], fkey(lg));
  }
}

// ---- pass 2: ex = exp(logit - m); denom += ex ----
__global__ void rgat_denom_kernel(const long long* __restrict__ eidx,
                                  float* __restrict__ logits,
                                  const unsigned* __restrict__ maxkey,
                                  float* __restrict__ denom) {
  int i = blockIdx.x * blockDim.x + threadIdx.x;
  if (i >= EEDG * HH) return;
  int e = i >> 2, h = i & 3;
  long long dst = eidx[EEDG + e];
  unsigned k = maxkey[(size_t)dst * HH + h];
  float m = (k == 0u) ? 0.0f : fkey_dec(k);
  float ex = __expf(logits[i] - m);
  logits[i] = ex;  // reuse buffer as ex
  atomicAdd(&denom[(size_t)dst * HH + h], ex);
}

// ---- pass 3: out[dst] += (v[src] + e_rel) * alpha (wave per edge) ----
__global__ __launch_bounds__(256)
void rgat_scatter_kernel(const float* __restrict__ V, const float* __restrict__ erel,
                         const long long* __restrict__ eidx,
                         const long long* __restrict__ etype,
                         const float* __restrict__ ex,
                         const float* __restrict__ denom,
                         float* __restrict__ out) {
  const int e    = blockIdx.x * 8 + (threadIdx.x >> 5);
  const int lane = threadIdx.x & 31;
  const long long src = eidx[e];
  const long long dst = eidx[EEDG + e];
  const long long rel = etype[e];
  const int h = lane >> 3;
  const float alpha = ex[(size_t)e * HH + h] /
                      (denom[(size_t)dst * HH + h] + 1e-16f);
  const float4 vv = *(const float4*)(V    + (size_t)src * HC + lane * 4);
  const float4 ev = *(const float4*)(erel + (size_t)rel * HC + lane * 4);
  float* op = out + (size_t)dst * HC + lane * 4;
  atomicAdd(op + 0, (vv.x + ev.x) * alpha);
  atomicAdd(op + 1, (vv.y + ev.y) * alpha);
  atomicAdd(op + 2, (vv.z + ev.z) * alpha);
  atomicAdd(op + 3, (vv.w + ev.w) * alpha);
}

extern "C" void kernel_launch(void* const* d_in, const int* in_sizes, int n_in,
                              void* d_out, int out_size, void* d_ws, size_t ws_size,
                              hipStream_t stream) {
  const float*     x     = (const float*)d_in[0];
  const long long* eidx  = (const long long*)d_in[1];  // [2,E] int64
  const long long* etype = (const long long*)d_in[2];  // [E]   int64
  const float*     emb   = (const float*)d_in[3];      // [R,DE]
  const float*     Wq    = (const float*)d_in[4];
  const float*     Wk    = (const float*)d_in[5];
  const float*     Wv    = (const float*)d_in[6];
  const float*     We    = (const float*)d_in[7];
  const float*     Wsk   = (const float*)d_in[8];
  float*           out   = (float*)d_out;

  float* ws      = (float*)d_ws;
  float* Q       = ws;                                  // N*128
  float* K       = Q    + (size_t)NN * HC;              // N*128
  float* V       = K    + (size_t)NN * HC;              // N*128
  float* erel    = V    + (size_t)NN * HC;              // R*128
  float* logits  = erel + (size_t)RRE * HC;             // E*H (logits -> ex)
  unsigned* maxkey = (unsigned*)(logits + (size_t)EEDG * HH);  // N*H
  float* denom   = (float*)(maxkey + (size_t)NN * HH);  // N*H
  _Float16* Wt   = (_Float16*)(denom + (size_t)NN * HH);// 4*128*128 f16

  rgat_init_kernel   <<<(NN * HH + 255) / 256, 256, 0, stream>>>(maxkey, denom);
  rgat_erel_kernel   <<<(RRE * HC + 255) / 256, 256, 0, stream>>>(emb, We, erel);
  rgat_wtconv_kernel <<<(4 * HC * FFE + 255) / 256, 256, 0, stream>>>(Wq, Wk, Wv, Wsk, Wt);
  rgat_proj_kernel   <<<NN / 16, 256, 0, stream>>>(x, Wt, Q, K, V, out);
  rgat_logits_kernel <<<EEDG / 8, 256, 0, stream>>>(Q, K, erel, eidx, etype, logits, maxkey);
  rgat_denom_kernel  <<<(EEDG * HH + 255) / 256, 256, 0, stream>>>(eidx, logits, maxkey, denom);
  rgat_scatter_kernel<<<EEDG / 8, 256, 0, stream>>>(V, erel, eidx, etype, logits, denom, out);
}
